// Decoder_56916906606847
// MI455X (gfx1250) — matrix operand order repaired and checked by reading, not compiled, not run on hardware
//
#include <hip/hip_runtime.h>
#include <hip/hip_bf16.h>
#include <stdint.h>

#define Bc 16
#define Tc 64
#define Sc 512
#define Vc 32000
#define Ec 512
#define Hc 1024
#define OOVc 30
#define VEXTc (Vc + OOVc)
#define NEG_INF (-1e12f)

typedef __bf16 bf16_t;
typedef __attribute__((ext_vector_type(16))) __bf16 v16bf;
typedef __attribute__((ext_vector_type(8)))  float  v8f;

union Frag { v16bf v; uint4 q[2]; };

__device__ inline bf16_t f2bf(float f) {
    union { float f; unsigned int u; } x; x.f = f;
    unsigned int r = (x.u + 0x7FFFu + ((x.u >> 16) & 1u)) >> 16;
    union { unsigned short s; bf16_t b; } y; y.s = (unsigned short)r;
    return y.b;
}

// A fragment: 16x32 bf16, lane L: M = L%16, half=L/16.
// elements 0..7  -> K = k0 + 8*half + j        (contiguous)
// elements 8..15 -> K = k0 + 16 + 8*half + j   (contiguous)
__device__ inline v16bf loadA(const bf16_t* __restrict__ A, int lda, int row, int k0, int half) {
    const bf16_t* p = A + (size_t)row * lda + k0 + 8 * half;
    Frag a;
    a.q[0] = *reinterpret_cast<const uint4*>(p);
    a.q[1] = *reinterpret_cast<const uint4*>(p + 16);
    return a.v;
}

// B fragment: 32x16 bf16 (K x N), B(k,n) = W[n,k] (W row-major [N,K]).
// lane L: N = n0 + L%16; elements 0..7 -> K = k0 + 8*half + j; elements 8..15 -> K = k0 + 16 + 8*half + j.
__device__ inline v16bf loadB(const bf16_t* __restrict__ W, int ldb, int col, int k0, int half) {
    const bf16_t* p = W + (size_t)col * ldb + k0 + 8 * half;
    Frag b;
    b.q[0] = *reinterpret_cast<const uint4*>(p);
    b.q[1] = *reinterpret_cast<const uint4*>(p + 16);
    return b.v;
}

__device__ inline v8f wmma_bf16(v16bf a, v16bf b, v8f c) {
    return __builtin_amdgcn_wmma_f32_16x16x32_bf16(false, a, false, b, (short)0, c, false, false);
}

// ---------------- conversion ----------------
__global__ void k_f2bf(const float* __restrict__ in, bf16_t* __restrict__ out, int n) {
    int i = blockIdx.x * blockDim.x + threadIdx.x;
    if (i < n) out[i] = f2bf(in[i]);
}

// ---------------- generic bf16 WMMA GEMM: D[M,N] = A[M,K] * W[N,K]^T + bias ----------------
__global__ void k_gemm(const bf16_t* __restrict__ A, int lda,
                       const bf16_t* __restrict__ W, int ldb,
                       const float* __restrict__ bias,
                       float* __restrict__ outf, int ldd,
                       bf16_t* __restrict__ outbf,
                       int N, int K, int m_tiles, int act) {
    int wave = (blockIdx.x * blockDim.x + threadIdx.x) >> 5;
    int lane = threadIdx.x & 31;
    int n_tiles = N >> 4;
    if (wave >= m_tiles * n_tiles) return;
    int mt = wave / n_tiles, nt = wave % n_tiles;
    int m0 = mt << 4, n0 = nt << 4;
    int half = lane >> 4;
    int mrow = m0 + (lane & 15);
    int ncol = n0 + (lane & 15);
    v8f c = {};
    for (int k0 = 0; k0 < K; k0 += 32) {
        v16bf a = loadA(A, lda, mrow, k0, half);
        v16bf b = loadB(W, ldb, ncol, k0, half);
        c = wmma_bf16(a, b, c);
    }
    float bv = bias ? bias[ncol] : 0.0f;
    for (int r = 0; r < 8; ++r) {
        int m = m0 + r + (half << 3);
        float v = c[r] + bv;
        if (act) v = tanhf(v);
        size_t idx = (size_t)m * ldd + ncol;
        outf[idx] = v;
        if (outbf) outbf[idx] = f2bf(v);
    }
}

// ---------------- gates = x @ W_ih^T + h @ W_hh^T + b_ih + b_hh ----------------
__global__ void k_gates(const bf16_t* __restrict__ xbf, const bf16_t* __restrict__ Wih,
                        const bf16_t* __restrict__ hbf, const bf16_t* __restrict__ Whh,
                        const float* __restrict__ bih, const float* __restrict__ bhh,
                        float* __restrict__ gates) {
    int wave = (blockIdx.x * blockDim.x + threadIdx.x) >> 5;
    int lane = threadIdx.x & 31;
    if (wave >= (4 * Hc) / 16) return;
    int n0 = wave << 4;
    int half = lane >> 4;
    int mrow = lane & 15;
    int ncol = n0 + (lane & 15);
    v8f c = {};
    for (int k0 = 0; k0 < Ec; k0 += 32)
        c = wmma_bf16(loadA(xbf, Ec, mrow, k0, half), loadB(Wih, Ec, ncol, k0, half), c);
    for (int k0 = 0; k0 < Hc; k0 += 32)
        c = wmma_bf16(loadA(hbf, Hc, mrow, k0, half), loadB(Whh, Hc, ncol, k0, half), c);
    float bv = bih[ncol] + bhh[ncol];
    for (int r = 0; r < 8; ++r) {
        int m = r + (half << 3);
        gates[(size_t)m * (4 * Hc) + ncol] = c[r] + bv;
    }
}

// ---------------- state init ----------------
__global__ void k_init_state(const float* __restrict__ h0, const float* __restrict__ c0,
                             float* __restrict__ h, float* __restrict__ c,
                             float* __restrict__ ctx, bf16_t* __restrict__ hbf) {
    int i = blockIdx.x * blockDim.x + threadIdx.x;
    if (i >= Bc * Hc) return;
    h[i] = h0[i]; c[i] = c0[i]; ctx[i] = 0.0f; hbf[i] = f2bf(h0[i]);
}

// ---------------- concat(emb[y_t], ctx) -> bf16 ----------------
__global__ void k_concat_red(const int* __restrict__ trg, const float* __restrict__ emb,
                             const float* __restrict__ ctx, bf16_t* __restrict__ out, int t) {
    int i = blockIdx.x * blockDim.x + threadIdx.x;
    const int Wd = Ec + Hc;
    if (i >= Bc * Wd) return;
    int b = i / Wd, j = i % Wd;
    float v;
    if (j < Ec) { int y = trg[b * Tc + t]; v = emb[(size_t)y * Ec + j]; }
    else        { v = ctx[b * Hc + (j - Ec)]; }
    out[i] = f2bf(v);
}

// ---------------- concat(h, ctx) -> bf16 ----------------
__global__ void k_concat_cat(const float* __restrict__ h, const float* __restrict__ ctx,
                             bf16_t* __restrict__ out) {
    int i = blockIdx.x * blockDim.x + threadIdx.x;
    if (i >= Bc * 2 * Hc) return;
    int b = i / (2 * Hc), j = i % (2 * Hc);
    float v = (j < Hc) ? h[b * Hc + j] : ctx[b * Hc + (j - Hc)];
    out[i] = f2bf(v);
}

// ---------------- LSTM cell pointwise ----------------
__global__ void k_lstm(const float* __restrict__ gates, float* __restrict__ c,
                       float* __restrict__ h, bf16_t* __restrict__ hbf) {
    int i = blockIdx.x * blockDim.x + threadIdx.x;
    if (i >= Bc * Hc) return;
    int b = i / Hc, j = i % Hc;
    const float* g = gates + (size_t)b * 4 * Hc;
    float ig = g[j], fg = g[Hc + j], gg = g[2 * Hc + j], og = g[3 * Hc + j];
    float si = 1.0f / (1.0f + expf(-ig));
    float sf = 1.0f / (1.0f + expf(-fg));
    float so = 1.0f / (1.0f + expf(-og));
    float cn = sf * c[i] + si * tanhf(gg);
    float hn = so * tanhf(cn);
    c[i] = cn; h[i] = hn; hbf[i] = f2bf(hn);
}

// ---------------- energy[b,s] = h . memories[b,s,:], masked ----------------
__global__ void k_energy(const float* __restrict__ h, const float* __restrict__ mem,
                         const unsigned char* __restrict__ mask, float* __restrict__ energy) {
    int wave = (blockIdx.x * blockDim.x + threadIdx.x) >> 5;
    int lane = threadIdx.x & 31;
    if (wave >= Bc * Sc) return;
    int b = wave / Sc, s = wave % Sc;
    const float* mp = mem + ((size_t)b * Sc + s) * Hc;
    const float* hp = h + (size_t)b * Hc;
    float acc = 0.0f;
    for (int i = lane; i < Hc; i += 32) acc += hp[i] * mp[i];
    for (int off = 16; off > 0; off >>= 1) acc += __shfl_xor(acc, off, 32);
    if (lane == 0) energy[wave] = mask[wave] ? NEG_INF : acc;
}

// ---------------- softmax over S per batch ----------------
__global__ void k_softmax(const float* __restrict__ energy, float* __restrict__ attn) {
    __shared__ float red[256];
    int b = blockIdx.x, tid = threadIdx.x;
    const float* e = energy + (size_t)b * Sc;
    float m = -3.4e38f;
    for (int s = tid; s < Sc; s += 256) m = fmaxf(m, e[s]);
    red[tid] = m; __syncthreads();
    for (int o = 128; o > 0; o >>= 1) { if (tid < o) red[tid] = fmaxf(red[tid], red[tid + o]); __syncthreads(); }
    float mx = red[0]; __syncthreads();
    float sum = 0.0f;
    for (int s = tid; s < Sc; s += 256) sum += expf(e[s] - mx);
    red[tid] = sum; __syncthreads();
    for (int o = 128; o > 0; o >>= 1) { if (tid < o) red[tid] += red[tid + o]; __syncthreads(); }
    float inv = 1.0f / red[0];
    for (int s = tid; s < Sc; s += 256) attn[(size_t)b * Sc + s] = expf(e[s] - mx) * inv;
}

// ---------------- ctx[b,h] = sum_s attn[b,s] * memories[b,s,h] ----------------
__global__ void k_ctx(const float* __restrict__ attn, const float* __restrict__ mem,
                      float* __restrict__ ctx) {
    int i = blockIdx.x * blockDim.x + threadIdx.x;
    if (i >= Bc * Hc) return;
    int b = i / Hc, j = i % Hc;
    const float* a = attn + (size_t)b * Sc;
    const float* mp = mem + (size_t)b * Sc * Hc + j;
    float acc = 0.0f;
    for (int s = 0; s < Sc; ++s) acc += a[s] * mp[(size_t)s * Hc];
    ctx[i] = acc;
}

// ---------------- per-step extended-vocab score buffer ----------------
__global__ void k_scores_init(float* __restrict__ scores) {
    int i = blockIdx.x * blockDim.x + threadIdx.x;
    if (i < Bc * VEXTc) scores[i] = NEG_INF;
}

__global__ void k_scatter(const int* __restrict__ esrc, const float* __restrict__ energy,
                          float* __restrict__ scores) {
    int i = blockIdx.x * blockDim.x + threadIdx.x;
    if (i >= Bc * Sc) return;
    int b = i / Sc;
    int v = esrc[i];
    float e = energy[i];
    unsigned int* addr = (unsigned int*)(scores + (size_t)b * VEXTc + v);
    unsigned int old = *addr;
    while (__uint_as_float(old) < e) {
        unsigned int assumed = old;
        old = atomicCAS(addr, assumed, __float_as_uint(e));
        if (old == assumed) break;
    }
}

__global__ void k_combine(const float* __restrict__ scores, float* __restrict__ dout, int t) {
    int i = blockIdx.x * blockDim.x + threadIdx.x;
    if (i >= Bc * VEXTc) return;
    int b = i / VEXTc, v = i % VEXTc;
    float adj = scores[i];
    adj = (adj == NEG_INF) ? 0.0f : adj;
    size_t off = ((size_t)b * Tc + t) * VEXTc + v;
    float base = (v < Vc) ? dout[off] : 0.0f;  // ext_logit zero-padded beyond V
    float r = base + adj;
    r = (r == 0.0f) ? NEG_INF : r;
    dout[off] = r;
}

extern "C" void kernel_launch(void* const* d_in, const int* in_sizes, int n_in,
                              void* d_out, int out_size, void* d_ws, size_t ws_size,
                              hipStream_t stream) {
    (void)in_sizes; (void)n_in; (void)out_size; (void)ws_size;
    const int*   trg   = (const int*)d_in[0];
    const int*   esrc  = (const int*)d_in[1];
    const float* h0    = (const float*)d_in[2];
    const float* c0    = (const float*)d_in[3];
    const float* enc   = (const float*)d_in[4];
    const unsigned char* mask = (const unsigned char*)d_in[5];
    const float* emb   = (const float*)d_in[7];
    const float* W_enc = (const float*)d_in[8];
    const float* b_enc = (const float*)d_in[9];
    const float* W_red = (const float*)d_in[10];
    const float* b_red = (const float*)d_in[11];
    const float* W_ih  = (const float*)d_in[12];
    const float* W_hh  = (const float*)d_in[13];
    const float* b_ih  = (const float*)d_in[14];
    const float* b_hh  = (const float*)d_in[15];
    const float* W_cat = (const float*)d_in[16];
    const float* b_cat = (const float*)d_in[17];
    const float* W_out = (const float*)d_in[18];
    const float* b_out = (const float*)d_in[19];
    float* out = (float*)d_out;

    char* ws = (char*)d_ws;
    size_t off = 0;
    auto alloc = [&](size_t bytes) -> void* {
        void* p = ws + off;
        off = (off + bytes + 255) & ~(size_t)255;
        return p;
    };

    bf16_t* wenc_bf = (bf16_t*)alloc((size_t)Hc * Hc * 2);
    bf16_t* wred_bf = (bf16_t*)alloc((size_t)Ec * (Ec + Hc) * 2);
    bf16_t* wih_bf  = (bf16_t*)alloc((size_t)4 * Hc * Ec * 2);
    bf16_t* whh_bf  = (bf16_t*)alloc((size_t)4 * Hc * Hc * 2);
    bf16_t* wcat_bf = (bf16_t*)alloc((size_t)Hc * 2 * Hc * 2);
    bf16_t* wout_bf = (bf16_t*)alloc((size_t)Vc * Hc * 2);
    bf16_t* enc_bf  = (bf16_t*)alloc((size_t)Bc * Sc * Hc * 2);
    float*  mem_f   = (float*)alloc((size_t)Bc * Sc * Hc * 4);
    float*  h_f     = (float*)alloc((size_t)Bc * Hc * 4);
    float*  c_f     = (float*)alloc((size_t)Bc * Hc * 4);
    float*  ctx_f   = (float*)alloc((size_t)Bc * Hc * 4);
    bf16_t* h_bf    = (bf16_t*)alloc((size_t)Bc * Hc * 2);
    bf16_t* ared_bf = (bf16_t*)alloc((size_t)Bc * (Ec + Hc) * 2);
    float*  x_f     = (float*)alloc((size_t)Bc * Ec * 4);
    bf16_t* x_bf    = (bf16_t*)alloc((size_t)Bc * Ec * 2);
    float*  gates_f = (float*)alloc((size_t)Bc * 4 * Hc * 4);
    float*  energy_f= (float*)alloc((size_t)Bc * Sc * 4);
    float*  attn_f  = (float*)alloc((size_t)Bc * Sc * 4);
    bf16_t* acat_bf = (bf16_t*)alloc((size_t)Bc * 2 * Hc * 2);
    float*  li_f    = (float*)alloc((size_t)Bc * Hc * 4);
    bf16_t* li_bf   = (bf16_t*)alloc((size_t)Bc * Hc * 2);
    float*  scores_f= (float*)alloc((size_t)Bc * VEXTc * 4);

    const int THR = 256;
    auto cdiv = [](int a, int b) { return (a + b - 1) / b; };

    // ---- one-time precompute ----
    k_f2bf<<<cdiv(Hc * Hc, THR), THR, 0, stream>>>(W_enc, wenc_bf, Hc * Hc);
    k_f2bf<<<cdiv(Ec * (Ec + Hc), THR), THR, 0, stream>>>(W_red, wred_bf, Ec * (Ec + Hc));
    k_f2bf<<<cdiv(4 * Hc * Ec, THR), THR, 0, stream>>>(W_ih, wih_bf, 4 * Hc * Ec);
    k_f2bf<<<cdiv(4 * Hc * Hc, THR), THR, 0, stream>>>(W_hh, whh_bf, 4 * Hc * Hc);
    k_f2bf<<<cdiv(Hc * 2 * Hc, THR), THR, 0, stream>>>(W_cat, wcat_bf, Hc * 2 * Hc);
    k_f2bf<<<cdiv(Vc * Hc, THR), THR, 0, stream>>>(W_out, wout_bf, Vc * Hc);
    k_f2bf<<<cdiv(Bc * Sc * Hc, THR), THR, 0, stream>>>(enc, enc_bf, Bc * Sc * Hc);

    // memories = enc @ W_enc^T + b_enc  : M=8192, N=1024, K=1024
    {
        int m_tiles = (Bc * Sc) / 16, n_tiles = Hc / 16;
        int waves = m_tiles * n_tiles;                 // 32768
        k_gemm<<<waves / 8, THR, 0, stream>>>(enc_bf, Hc, wenc_bf, Hc, b_enc,
                                              mem_f, Hc, (bf16_t*)nullptr,
                                              Hc, Hc, m_tiles, 0);
    }
    k_init_state<<<cdiv(Bc * Hc, THR), THR, 0, stream>>>(h0, c0, h_f, c_f, ctx_f, h_bf);

    // ---- sequential decode steps ----
    for (int t = 0; t < Tc; ++t) {
        k_concat_red<<<cdiv(Bc * (Ec + Hc), THR), THR, 0, stream>>>(trg, emb, ctx_f, ared_bf, t);
        // x = Ared @ W_red^T + b_red : N=512, K=1536 -> 32 waves
        k_gemm<<<4, THR, 0, stream>>>(ared_bf, Ec + Hc, wred_bf, Ec + Hc, b_red,
                                      x_f, Ec, x_bf, Ec, Ec + Hc, 1, 0);
        // gates : N=4096, K=512 + K=1024 -> 256 waves
        k_gates<<<32, THR, 0, stream>>>(x_bf, wih_bf, h_bf, whh_bf, b_ih, b_hh, gates_f);
        k_lstm<<<cdiv(Bc * Hc, THR), THR, 0, stream>>>(gates_f, c_f, h_f, h_bf);
        // energy : one wave per (b,s) -> 8192 waves
        k_energy<<<(Bc * Sc) / 8, THR, 0, stream>>>(h_f, mem_f, mask, energy_f);
        k_softmax<<<Bc, THR, 0, stream>>>(energy_f, attn_f);
        k_ctx<<<cdiv(Bc * Hc, THR), THR, 0, stream>>>(attn_f, mem_f, ctx_f);
        k_concat_cat<<<cdiv(Bc * 2 * Hc, THR), THR, 0, stream>>>(h_f, ctx_f, acat_bf);
        // logit_in = tanh(Acat @ W_cat^T + b_cat) : N=1024, K=2048 -> 64 waves
        k_gemm<<<8, THR, 0, stream>>>(acat_bf, 2 * Hc, wcat_bf, 2 * Hc, b_cat,
                                      li_f, Hc, li_bf, Hc, 2 * Hc, 1, 1);
        // logit = logit_in @ W_out^T + b_out -> directly into d_out[:, t, :V]
        // N=32000, K=1024 -> 2000 waves, out row stride = T*VEXT
        k_gemm<<<250, THR, 0, stream>>>(li_bf, Hc, wout_bf, Hc, b_out,
                                        out + (size_t)t * VEXTc, Tc * VEXTc,
                                        (bf16_t*)nullptr, Vc, Hc, 1, 0);
        // pointer scatter-max and final combine
        k_scores_init<<<cdiv(Bc * VEXTc, THR), THR, 0, stream>>>(scores_f);
        k_scatter<<<cdiv(Bc * Sc, THR), THR, 0, stream>>>(esrc, energy_f, scores_f);
        k_combine<<<cdiv(Bc * VEXTc, THR), THR, 0, stream>>>(scores_f, out, t);
    }
}
